// InteractionEncoder_51041391346020
// MI455X (gfx1250) — compile-verified, matrix operand order checked by reading, not run on hardware
//
#include <hip/hip_runtime.h>

// ---------------------------------------------------------------- types
typedef __attribute__((ext_vector_type(16))) __bf16 v16bf;
typedef __attribute__((ext_vector_type(8)))  __bf16 v8bf;
typedef __attribute__((ext_vector_type(8)))  float  v8f;

#define NN   2048   // B*NA nodes
#define DD   128
#define HH   6
#define HD   768    // H*D

// ---------------------------------------------------------------- helpers
__device__ __forceinline__ __bf16 f2bf(float f) {
  union { float f; unsigned u; } x; x.f = f;
  unsigned r = x.u + 0x7FFFu + ((x.u >> 16) & 1u);   // round-to-nearest-even
  union { unsigned short s; __bf16 b; } y; y.s = (unsigned short)(r >> 16);
  return y.b;
}

// Load one 16x32 bf16 WMMA operand fragment (A or B layout, both identical):
// lane covers row (m or n) = lane&15; klo = lane<16 ? 0 : 8;
// elements 0-7  = K[klo .. klo+7]   (one b128)
// elements 8-15 = K[klo+16..klo+23] (one b128)
// `p` must already point at  base + (lane&15)*ld + klo.
__device__ __forceinline__ v16bf load_frag(const __bf16* p) {
  v8bf lo = *(const v8bf*)(p);
  v8bf hi = *(const v8bf*)(p + 16);
  return __builtin_shufflevector(lo, hi, 0,1,2,3,4,5,6,7,8,9,10,11,12,13,14,15);
}

__device__ __forceinline__ v8f wmma_bf16(v16bf a, v16bf b, v8f c) {
  return __builtin_amdgcn_wmma_f32_16x16x32_bf16(
      false, a, false, b, (short)0, c, false, false);
}

// ---------------------------------------------------------------- f32 -> bf16
__global__ void cvt_bf16(const float* __restrict__ x, __bf16* __restrict__ y, int n) {
  int i = blockIdx.x * blockDim.x + threadIdx.x;
  if (i < n) y[i] = f2bf(x[i]);
}

// ---------------------------------------------------------------- WMMA GEMM tile body
// C[16 x 64 strip] = act( A[M,K] * W[N,K]^T  (+ R[M,N]) ), one wave.
__device__ __forceinline__
void gemm_tile(const __bf16* __restrict__ A, const __bf16* __restrict__ W,
               const float* __restrict__ R, float* __restrict__ C,
               int m0, int n0, int N, int K, int relu_flag, int lane) {
  const int klo = (lane < 16) ? 0 : 8;
  const int rsel = lane & 15;

  const __bf16* ap  = A + (size_t)(m0 + rsel) * K + klo;
  const __bf16* wp0 = W + (size_t)(n0 +  0 + rsel) * K + klo;
  const __bf16* wp1 = W + (size_t)(n0 + 16 + rsel) * K + klo;
  const __bf16* wp2 = W + (size_t)(n0 + 32 + rsel) * K + klo;
  const __bf16* wp3 = W + (size_t)(n0 + 48 + rsel) * K + klo;

  // pull panels toward the WGP while first fragments load (global_prefetch_b8)
  __builtin_prefetch(ap,  0, 3);
  __builtin_prefetch(wp0, 0, 3);
  __builtin_prefetch(wp1, 0, 3);
  __builtin_prefetch(wp2, 0, 3);
  __builtin_prefetch(wp3, 0, 3);

  v8f acc0 = {}, acc1 = {}, acc2 = {}, acc3 = {};
  for (int k = 0; k < K; k += 32) {
    v16bf af = load_frag(ap + k);
    acc0 = wmma_bf16(af, load_frag(wp0 + k), acc0);
    acc1 = wmma_bf16(af, load_frag(wp1 + k), acc1);
    acc2 = wmma_bf16(af, load_frag(wp2 + k), acc2);
    acc3 = wmma_bf16(af, load_frag(wp3 + k), acc3);
  }

  const int col = lane & 15;
  const int rb  = (lane < 16) ? 0 : 8;     // C/D layout: VGPR r -> row r (+8 hi half)
  v8f accs[4] = {acc0, acc1, acc2, acc3};
#pragma unroll
  for (int t = 0; t < 4; ++t) {
#pragma unroll
    for (int r = 0; r < 8; ++r) {
      size_t idx = (size_t)(m0 + rb + r) * N + (n0 + t * 16 + col);
      float v = accs[t][r];
      if (R) v += R[idx];
      if (relu_flag) v = fmaxf(v, 0.f);
      C[idx] = v;
    }
  }
}

// generic GEMM: grid = (M/16, N/64); 1 wave per block.
__global__ __launch_bounds__(32)
void gemm_wmma(const __bf16* __restrict__ A, const __bf16* __restrict__ W,
               const float* __restrict__ R, float* __restrict__ C,
               int M, int N, int K, int relu_flag) {
  gemm_tile(A, W, R, C, blockIdx.x * 16, blockIdx.y * 64, N, K, relu_flag,
            threadIdx.x);
}

// fused QKV projections: blockIdx.z selects which weight/output.
__global__ __launch_bounds__(32)
void gemm_qkv(const __bf16* __restrict__ A,
              const __bf16* __restrict__ Wq, const __bf16* __restrict__ Wk,
              const __bf16* __restrict__ Wv,
              float* __restrict__ Cq, float* __restrict__ Ck,
              float* __restrict__ Cv, int N, int K) {
  const __bf16* W = (blockIdx.z == 0) ? Wq : (blockIdx.z == 1) ? Wk : Wv;
  float*        C = (blockIdx.z == 0) ? Cq : (blockIdx.z == 1) ? Ck : Cv;
  gemm_tile(A, W, nullptr, C, blockIdx.x * 16, blockIdx.y * 64, N, K, 0,
            threadIdx.x);
}

// ---------------------------------------------------------------- LayerNorm(768) -> bf16
// grid.y selects (Q,K,V); relu only for V.
__global__ __launch_bounds__(256)
void ln768_qkv(const float* __restrict__ Xq, const float* __restrict__ Xk,
               const float* __restrict__ Xv,
               const float* __restrict__ gq, const float* __restrict__ bq,
               const float* __restrict__ gk, const float* __restrict__ bk,
               const float* __restrict__ gv, const float* __restrict__ bv,
               __bf16* __restrict__ Yq, __bf16* __restrict__ Yk,
               __bf16* __restrict__ Yv) {
  const int which = blockIdx.y;
  const float* X = (which == 0) ? Xq : (which == 1) ? Xk : Xv;
  const float* g = (which == 0) ? gq : (which == 1) ? gk : gv;
  const float* b = (which == 0) ? bq : (which == 1) ? bk : bv;
  __bf16*      Y = (which == 0) ? Yq : (which == 1) ? Yk : Yv;
  const int relu_flag = (which == 2);

  const int wave = threadIdx.x >> 5, lane = threadIdx.x & 31;
  const int row = blockIdx.x * 8 + wave;
  const float* x = X + (size_t)row * HD;
  float vals[24], s = 0.f, s2 = 0.f;
#pragma unroll
  for (int i = 0; i < 24; ++i) {
    float f = x[lane + 32 * i];
    vals[i] = f; s += f; s2 += f * f;
  }
#pragma unroll
  for (int off = 16; off; off >>= 1) {
    s  += __shfl_xor(s,  off, 32);
    s2 += __shfl_xor(s2, off, 32);
  }
  float mean = s * (1.f / HD);
  float inv  = rsqrtf(s2 * (1.f / HD) - mean * mean + 1e-5f);
#pragma unroll
  for (int i = 0; i < 24; ++i) {
    int c = lane + 32 * i;
    float y = (vals[i] - mean) * inv * g[c] + b[c];
    if (relu_flag) y = fmaxf(y, 0.f);
    Y[(size_t)row * HD + c] = f2bf(y);
  }
}

// ---------------------------------------------------------------- LayerNorm(128) -> bf16
// X2 optional second addend (pre-LN residual).
__global__ __launch_bounds__(256)
void ln128(const float* __restrict__ X, const float* __restrict__ X2,
           const float* __restrict__ g, const float* __restrict__ b,
           __bf16* __restrict__ Y, int relu_flag) {
  const int wave = threadIdx.x >> 5, lane = threadIdx.x & 31;
  const int row = blockIdx.x * 8 + wave;
  const size_t base = (size_t)row * DD;
  float vals[4], s = 0.f, s2 = 0.f;
#pragma unroll
  for (int i = 0; i < 4; ++i) {
    int c = lane + 32 * i;
    float f = X[base + c];
    if (X2) f += X2[base + c];
    vals[i] = f; s += f; s2 += f * f;
  }
#pragma unroll
  for (int off = 16; off; off >>= 1) {
    s  += __shfl_xor(s,  off, 32);
    s2 += __shfl_xor(s2, off, 32);
  }
  float mean = s * (1.f / DD);
  float inv  = rsqrtf(s2 * (1.f / DD) - mean * mean + 1e-5f);
#pragma unroll
  for (int i = 0; i < 4; ++i) {
    int c = lane + 32 * i;
    float y = (vals[i] - mean) * inv * g[c] + b[c];
    if (relu_flag) y = fmaxf(y, 0.f);
    Y[base + c] = f2bf(y);
  }
}

// ---------------------------------------------------------------- attention per (scene, head)
// Q,K,V: bf16 [NN, HD].  block = 128 threads (4 waves), grid = B*H = 384.
// S = Q Kt / sqrt(D); per-row softmax (global-max form cancels); O = P V -> bf16.
__global__ __launch_bounds__(128)
void attn(const __bf16* __restrict__ Q, const __bf16* __restrict__ Kb,
          const __bf16* __restrict__ V, __bf16* __restrict__ O) {
  __shared__ float  Sm[32 * 32];
  __shared__ __bf16 Pb[32 * 32];       // softmax probs, row-major
  __shared__ __bf16 VT[128 * 32];      // V transposed: VT[d][j]
  const int bh = blockIdx.x;
  const int sc = bh / HH, h = bh % HH;
  const int tid = threadIdx.x, wave = tid >> 5, lane = tid & 31;
  const size_t base = (size_t)(sc * 32) * HD + h * DD;

  // stage V^T into LDS
  for (int idx = tid; idx < 32 * 128; idx += 128) {
    int j = idx >> 7, d = idx & 127;
    VT[d * 32 + j] = V[base + (size_t)j * HD + d];
  }
  __syncthreads();

  const int klo = (lane < 16) ? 0 : 8;
  const int rsel = lane & 15;
  const int col = lane & 15;
  const int rb = (lane < 16) ? 0 : 8;

  // scores: 4 waves x one 16x16 tile, K=128 -> 4 WMMAs each
  {
    const int mt = wave >> 1, nt = wave & 1;
    v8f acc = {};
    const __bf16* qa = Q  + base + (size_t)(mt * 16 + rsel) * HD + klo;
    const __bf16* kb = Kb + base + (size_t)(nt * 16 + rsel) * HD + klo;
    __builtin_prefetch(qa, 0, 3);
    __builtin_prefetch(kb, 0, 3);
    for (int k = 0; k < DD; k += 32)
      acc = wmma_bf16(load_frag(qa + k), load_frag(kb + k), acc);
    const float scale = 0.08838834764831845f;   // 1/sqrt(128)
#pragma unroll
    for (int r = 0; r < 8; ++r)
      Sm[(mt * 16 + rb + r) * 32 + nt * 16 + col] = acc[r] * scale;
  }
  __syncthreads();

  // row softmax (exp(a - gmax)/segsum == per-row softmax: gmax cancels)
  if (tid < 32) {
    float rv[32], mx = -3.0e38f;
#pragma unroll
    for (int j = 0; j < 32; ++j) { rv[j] = Sm[tid * 32 + j]; mx = fmaxf(mx, rv[j]); }
    float sum = 0.f;
#pragma unroll
    for (int j = 0; j < 32; ++j) { rv[j] = __expf(rv[j] - mx); sum += rv[j]; }
    float inv = 1.f / sum;
#pragma unroll
    for (int j = 0; j < 32; ++j) Pb[tid * 32 + j] = f2bf(rv[j] * inv);
  }
  __syncthreads();

  // O = P[32x32] @ V[32x128]: 2 mtiles x 8 ntiles, K=32 -> 1 WMMA per tile.
  {
    const int mt = wave >> 1;
    const int ntb = (wave & 1) * 4;
    v16bf a = load_frag(&Pb[(mt * 16 + rsel) * 32 + klo]);
#pragma unroll
    for (int nn = 0; nn < 4; ++nn) {
      int nt = ntb + nn;
      v16bf bfr = load_frag(&VT[(nt * 16 + rsel) * 32 + klo]);
      v8f acc = {};
      acc = wmma_bf16(a, bfr, acc);
#pragma unroll
      for (int r = 0; r < 8; ++r)
        O[base + (size_t)(mt * 16 + rb + r) * HD + nt * 16 + col] = f2bf(acc[r]);
    }
  }
}

// ---------------------------------------------------------------- host
extern "C" void kernel_launch(void* const* d_in, const int* in_sizes, int n_in,
                              void* d_out, int out_size, void* d_ws, size_t ws_size,
                              hipStream_t stream) {
  (void)in_sizes; (void)n_in; (void)out_size; (void)ws_size;
  const float* agents = (const float*)d_in[0];
  // d_in[1] = hi, d_in[2] = wi : edge structure is fully-connected per scene; not needed.
  const float* Wq  = (const float*)d_in[3];
  const float* gq  = (const float*)d_in[4];
  const float* bq  = (const float*)d_in[5];
  const float* Wk  = (const float*)d_in[6];
  const float* gk  = (const float*)d_in[7];
  const float* bk  = (const float*)d_in[8];
  const float* Wv  = (const float*)d_in[9];
  const float* gv  = (const float*)d_in[10];
  const float* bv  = (const float*)d_in[11];
  const float* Wo1 = (const float*)d_in[12];
  const float* go  = (const float*)d_in[13];
  const float* bo  = (const float*)d_in[14];
  const float* Wo2 = (const float*)d_in[15];
  const float* W1  = (const float*)d_in[16];
  const float* gn  = (const float*)d_in[17];
  const float* bn  = (const float*)d_in[18];
  const float* W2  = (const float*)d_in[19];
  float* out = (float*)d_out;

  size_t off = 0;
  auto alloc = [&](size_t bytes) -> void* {
    void* p = (char*)d_ws + off;
    off += (bytes + 255) & ~(size_t)255;
    return p;
  };
  __bf16* Abf   = (__bf16*)alloc((size_t)NN * DD * 2);
  __bf16* Wqb   = (__bf16*)alloc((size_t)HD * DD * 2);
  __bf16* Wkb   = (__bf16*)alloc((size_t)HD * DD * 2);
  __bf16* Wvb   = (__bf16*)alloc((size_t)HD * DD * 2);
  __bf16* Wo1b  = (__bf16*)alloc((size_t)DD * HD * 2);
  __bf16* Wo2b  = (__bf16*)alloc((size_t)DD * DD * 2);
  __bf16* W1b   = (__bf16*)alloc((size_t)DD * DD * 2);
  __bf16* W2b   = (__bf16*)alloc((size_t)DD * DD * 2);
  float*  Qpre  = (float*) alloc((size_t)NN * HD * 4);
  float*  Kpre  = (float*) alloc((size_t)NN * HD * 4);
  float*  Vpre  = (float*) alloc((size_t)NN * HD * 4);
  __bf16* Qbf   = (__bf16*)alloc((size_t)NN * HD * 2);
  __bf16* Kbf   = (__bf16*)alloc((size_t)NN * HD * 2);
  __bf16* Vbf   = (__bf16*)alloc((size_t)NN * HD * 2);
  __bf16* Obf   = (__bf16*)alloc((size_t)NN * HD * 2);
  float*  T1    = (float*) alloc((size_t)NN * DD * 4);
  __bf16* Ubf   = (__bf16*)alloc((size_t)NN * DD * 2);
  float*  OutP  = (float*) alloc((size_t)NN * DD * 4);
  float*  N0    = (float*) alloc((size_t)NN * DD * 4);
  __bf16* Hbf   = (__bf16*)alloc((size_t)NN * DD * 2);

  auto cvt = [&](const float* x, __bf16* y, int n) {
    cvt_bf16<<<dim3((n + 255) / 256), dim3(256), 0, stream>>>(x, y, n);
  };
  cvt(agents, Abf, NN * DD);
  cvt(Wq,  Wqb,  HD * DD);
  cvt(Wk,  Wkb,  HD * DD);
  cvt(Wv,  Wvb,  HD * DD);
  cvt(Wo1, Wo1b, DD * HD);
  cvt(Wo2, Wo2b, DD * DD);
  cvt(W1,  W1b,  DD * DD);
  cvt(W2,  W2b,  DD * DD);

  dim3 blk32(32), blk128(128), blk256(256);
  // QKV projections (fused, z selects): [2048,128] x [128,768]
  gemm_qkv<<<dim3(NN / 16, HD / 64, 3), blk32, 0, stream>>>(
      Abf, Wqb, Wkb, Wvb, Qpre, Kpre, Vpre, HD, DD);
  // LayerNorm over 768 (fused Q/K/V; ReLU for V), emit bf16
  ln768_qkv<<<dim3(NN / 8, 3), blk256, 0, stream>>>(
      Qpre, Kpre, Vpre, gq, bq, gk, bk, gv, bv, Qbf, Kbf, Vbf);
  // per-(scene,head) attention
  attn<<<dim3((NN / 32) * HH), blk128, 0, stream>>>(Qbf, Kbf, Vbf, Obf);
  // to_out: LN(O @ Wo1^T) -> relu -> @ Wo2^T
  gemm_wmma<<<dim3(NN / 16, DD / 64), blk32, 0, stream>>>(Obf, Wo1b, nullptr, T1, NN, DD, HD, 0);
  ln128<<<dim3(NN / 8), blk256, 0, stream>>>(T1, nullptr, go, bo, Ubf, 1);
  gemm_wmma<<<dim3(NN / 16, DD / 64), blk32, 0, stream>>>(Ubf, Wo2b, nullptr, OutP, NN, DD, DD, 0);
  // node MLP: relu(LN(A@W1^T + out)) @ W2^T + agents, relu
  gemm_wmma<<<dim3(NN / 16, DD / 64), blk32, 0, stream>>>(Abf, W1b, nullptr, N0, NN, DD, DD, 0);
  ln128<<<dim3(NN / 8), blk256, 0, stream>>>(N0, OutP, gn, bn, Hbf, 1);
  gemm_wmma<<<dim3(NN / 16, DD / 64), blk32, 0, stream>>>(Hbf, W2b, agents, out, NN, DD, DD, 1);
}
